// Injector_27908697489888
// MI455X (gfx1250) — compile-verified, matrix operand order checked by reading, not run on hardware
//
#include <hip/hip_runtime.h>

typedef unsigned int u32;
typedef __attribute__((ext_vector_type(16))) __bf16 bf16x16;
typedef __attribute__((ext_vector_type(8)))  float  v8f;

union Frag { bf16x16 v; u32 u[8]; };

#define BSZ   2
#define NQX   4096
#define DIMX  768
#define HEADSX 6
#define HDX   128
#define NVX   4096
#define ROWS  8192          // BSZ*NQX == BSZ*NVX
#define KPACK 384           // DIMX/2 (bf16 pairs per row)

// round-to-nearest-even f32 -> bf16 pair pack
__device__ __forceinline__ u32 pack_bf16x2(float a, float b) {
  u32 ua = __builtin_bit_cast(u32, a);
  u32 ub = __builtin_bit_cast(u32, b);
  ua += 0x7FFFu + ((ua >> 16) & 1u);
  ub += 0x7FFFu + ((ub >> 16) & 1u);
  return ((ua >> 16) & 0xFFFFu) | (ub & 0xFFFF0000u);
}

// ---------------------------------------------------------------------------
// Tiled bf16 WMMA GEMM: C[M,N] = A[M,K] * B[K,N] (+bias), optional residual:
//   C = resid + gamma * (A*B + bias)
// A packed as [M][K/2] u32 (bf16 pairs, K-major). B packed as [N][K/2] u32.
// Block tile 128x128, K-step 32. 256 threads = 8 wave32 waves, wave grid 4x2,
// each wave computes 32x64 via 2x4 v_wmma_f32_16x16x32_bf16 fragments.
// ---------------------------------------------------------------------------
#define BM 128
#define BN 128
#define LDT 17   // padded LDS row stride (u32) to spread banks

__global__ __launch_bounds__(256) void gemm_bf16(
    const u32* __restrict__ Ap, const u32* __restrict__ Bp,
    const float* __restrict__ bias, float* __restrict__ C,
    const float* __restrict__ resid, const float* __restrict__ gamma,
    int M, int N, int K) {
  const int KP   = K >> 1;
  const int tid  = threadIdx.x;
  const int lane = tid & 31, wave = tid >> 5;
  const int wm = wave & 3, wn = wave >> 2;
  const int half = lane >> 4, ml = lane & 15;
  const int m0 = blockIdx.y * BM, n0 = blockIdx.x * BN;

  __shared__ u32 sA[BM * LDT];
  __shared__ u32 sB[BN * LDT];

  v8f acc[2][4];
  const v8f vz = {0.f,0.f,0.f,0.f,0.f,0.f,0.f,0.f};
#pragma unroll
  for (int f = 0; f < 2; ++f)
#pragma unroll
    for (int g = 0; g < 4; ++g) acc[f][g] = vz;

  const int nK = K / 32;
  for (int ks = 0; ks < nK; ++ks) {
    const int kp0 = ks * 16;
    // stage A/B tiles into LDS (each thread: 2 uint4 for A, 2 for B)
#pragma unroll
    for (int i = 0; i < 2; ++i) {
      int idx = tid + i * 256;          // 512 uint4 slots per tile
      int row = idx >> 2, c4 = (idx & 3) << 2;
      const uint4 va = *reinterpret_cast<const uint4*>(
          Ap + (size_t)(m0 + row) * KP + kp0 + c4);
      int sa = row * LDT + c4;
      sA[sa+0]=va.x; sA[sa+1]=va.y; sA[sa+2]=va.z; sA[sa+3]=va.w;
      uint4 vb = make_uint4(0u,0u,0u,0u);
      int brow = n0 + row;
      if (brow < N)
        vb = *reinterpret_cast<const uint4*>(Bp + (size_t)brow * KP + kp0 + c4);
      int sb = row * LDT + c4;
      sB[sb+0]=vb.x; sB[sb+1]=vb.y; sB[sb+2]=vb.z; sB[sb+3]=vb.w;
    }
    // gfx1250 speculative prefetch of the next A tile (global_prefetch_b8)
    if (ks + 1 < nK)
      __builtin_prefetch(Ap + (size_t)(m0 + (tid >> 1)) * KP + kp0 + 16
                             + ((tid & 1) << 3), 0, 0);
    __syncthreads();

    // fragment gathers per documented 16-bit A/B VGPR layouts:
    // lane = M (A) / N (B) index, u32 slot p holds K-pair kp = 4*half+p (+4 if p>=4)
    Frag a[2], b[4];
#pragma unroll
    for (int f = 0; f < 2; ++f) {
      int row = wm * 32 + f * 16 + ml;
#pragma unroll
      for (int p = 0; p < 8; ++p) {
        int kp = 4 * half + p + ((p < 4) ? 0 : 4);
        a[f].u[p] = sA[row * LDT + kp];
      }
    }
#pragma unroll
    for (int g = 0; g < 4; ++g) {
      int col = wn * 64 + g * 16 + ml;
#pragma unroll
      for (int p = 0; p < 8; ++p) {
        int kp = 4 * half + p + ((p < 4) ? 0 : 4);
        b[g].u[p] = sB[col * LDT + kp];
      }
    }
#pragma unroll
    for (int f = 0; f < 2; ++f)
#pragma unroll
      for (int g = 0; g < 4; ++g)
        acc[f][g] = __builtin_amdgcn_wmma_f32_16x16x32_bf16(
            false, a[f].v, false, b[g].v, (short)0, acc[f][g], false, false);
    __syncthreads();
  }

  // epilogue: D layout — VGPR r: lanes0-15 M=r, lanes16-31 M=r+8; N = lane&15
#pragma unroll
  for (int f = 0; f < 2; ++f) {
    int rowbase = m0 + wm * 32 + f * 16 + 8 * half;
#pragma unroll
    for (int g = 0; g < 4; ++g) {
      int col = n0 + wn * 64 + g * 16 + ml;
      if (col < N) {
#pragma unroll
        for (int r = 0; r < 8; ++r) {
          size_t o = (size_t)(rowbase + r) * N + col;
          float v = acc[f][g][r] + bias[col];
          if (resid) v = resid[o] + gamma[col] * v;
          C[o] = v;
        }
      }
    }
  }
}

// ---------------------------------------------------------------------------
// LayerNorm over DIMX, emit packed bf16 pairs (feeds WMMA GEMM A operand)
// ---------------------------------------------------------------------------
__global__ __launch_bounds__(128) void layernorm_pack(
    const float* __restrict__ x, const float* __restrict__ w,
    const float* __restrict__ b, u32* __restrict__ qp) {
  const int row = blockIdx.x, tid = threadIdx.x;
  const float* xr = x + (size_t)row * DIMX;
  float v0[3], v1[3], s = 0.f, sq = 0.f;
#pragma unroll
  for (int j = 0; j < 3; ++j) {
    int c = 2 * (tid + j * 128);
    v0[j] = xr[c]; v1[j] = xr[c + 1];
    s += v0[j] + v1[j];
    sq += v0[j] * v0[j] + v1[j] * v1[j];
  }
  __shared__ float rs[128], rq[128];
  rs[tid] = s; rq[tid] = sq; __syncthreads();
  for (int off = 64; off > 0; off >>= 1) {
    if (tid < off) { rs[tid] += rs[tid + off]; rq[tid] += rq[tid + off]; }
    __syncthreads();
  }
  const float mu   = rs[0] * (1.f / DIMX);
  const float var  = rq[0] * (1.f / DIMX) - mu * mu;
  const float rstd = rsqrtf(var + 1e-6f);
#pragma unroll
  for (int j = 0; j < 3; ++j) {
    int pi = tid + j * 128, c = 2 * pi;
    float y0 = (v0[j] - mu) * rstd * w[c]     + b[c];
    float y1 = (v1[j] - mu) * rstd * w[c + 1] + b[c + 1];
    qp[(size_t)row * KPACK + pi] = pack_bf16x2(y0, y1);
  }
}

// f32 row-major -> packed bf16 pairs (same layout, K-major)
__global__ void pack_rows(const float* __restrict__ src, u32* __restrict__ dst,
                          int n) {
  int i = blockIdx.x * blockDim.x + threadIdx.x;
  if (i < n) dst[i] = pack_bf16x2(src[2 * i], src[2 * i + 1]);
}

// W[K,N] f32 -> dst[(n_off+n)][K/2] packed bf16 pairs (N-major, for B operand)
__global__ void pack_weights(const float* __restrict__ W, u32* __restrict__ dst,
                             int N, int n_off) {
  int i = blockIdx.x * blockDim.x + threadIdx.x;
  if (i >= N * KPACK) return;
  int n = i / KPACK, kp = i % KPACK;
  dst[(size_t)(n_off + n) * KPACK + kp] =
      pack_bf16x2(W[(size_t)(2 * kp) * N + n], W[(size_t)(2 * kp + 1) * N + n]);
}

__global__ void concat_bias(const float* __restrict__ ob,
                            const float* __restrict__ ab,
                            float* __restrict__ dst) {
  int i = threadIdx.x;
  if (i < 48) dst[i] = ob[i];
  else if (i < 72) dst[i] = ab[i - 48];
}

// ---------------------------------------------------------------------------
// MSDA sampling: per query block. Joint softmax over 24 (heads*points) logits
// (matches reference axis=-1 pre-reshape), bilinear zero-padded gather on the
// 64x64xHEADSxHD value image, writes attn rows as packed bf16 for GEMM3.
// ---------------------------------------------------------------------------
__global__ __launch_bounds__(128) void msda_sample(
    const float* __restrict__ value, const float* __restrict__ qo,
    const float* __restrict__ refp, u32* __restrict__ attn) {
  const int r = blockIdx.x, tid = threadIdx.x;
  const int bb = r >> 12;                  // / NQX
  const float* qrow = qo + (size_t)r * 72;
  __shared__ float s_off[48], s_lg[24];
  if (tid < 48) s_off[tid] = qrow[tid];
  if (tid < 24) s_lg[tid] = qrow[48 + tid];
  __syncthreads();
  float m = -1e30f;
  for (int i = 0; i < 24; ++i) m = fmaxf(m, s_lg[i]);
  float sum = 0.f;
  for (int i = 0; i < 24; ++i) sum += expf(s_lg[i] - m);
  const float inv = 1.f / sum;
  const float rx = refp[(size_t)r * 2], ry = refp[(size_t)r * 2 + 1];
  const float* vb = value + (size_t)bb * NVX * DIMX;

#pragma unroll
  for (int j = 0; j < 3; ++j) {
    int pi = tid + j * 128;
    int c0 = pi * 2;
    int head = c0 >> 7, d = c0 & 127;
    float a0 = 0.f, a1 = 0.f;
#pragma unroll
    for (int p = 0; p < 4; ++p) {
      float ox = s_off[head * 8 + p * 2], oy = s_off[head * 8 + p * 2 + 1];
      // loc = ref + off/64 ; x = loc*64 - 0.5
      float xx = rx * 64.f + ox - 0.5f;
      float yy = ry * 64.f + oy - 0.5f;
      float xf = floorf(xx), yf = floorf(yy);
      float wx = xx - xf, wy = yy - yf;
      int x0 = (int)xf, y0 = (int)yf;
      float wgt = expf(s_lg[head * 4 + p] - m) * inv;
      auto g2 = [&](int yi, int xi, float& gx, float& gy) {
        gx = 0.f; gy = 0.f;
        if (xi >= 0 && xi < 64 && yi >= 0 && yi < 64) {
          const float2 t = *reinterpret_cast<const float2*>(
              vb + (size_t)(yi * 64 + xi) * DIMX + head * HDX + d);
          gx = t.x; gy = t.y;
        }
      };
      float v00x, v00y, v01x, v01y, v10x, v10y, v11x, v11y;
      g2(y0,     x0,     v00x, v00y);
      g2(y0,     x0 + 1, v01x, v01y);
      g2(y0 + 1, x0,     v10x, v10y);
      g2(y0 + 1, x0 + 1, v11x, v11y);
      float bl0 = (1.f - wy) * ((1.f - wx) * v00x + wx * v01x)
                +        wy  * ((1.f - wx) * v10x + wx * v11x);
      float bl1 = (1.f - wy) * ((1.f - wx) * v00y + wx * v01y)
                +        wy  * ((1.f - wx) * v10y + wx * v11y);
      a0 += wgt * bl0;
      a1 += wgt * bl1;
    }
    attn[(size_t)r * KPACK + pi] = pack_bf16x2(a0, a1);
  }
}

// ---------------------------------------------------------------------------
extern "C" void kernel_launch(void* const* d_in, const int* in_sizes, int n_in,
                              void* d_out, int out_size, void* d_ws,
                              size_t ws_size, hipStream_t stream) {
  const float* query = (const float*)d_in[0];
  const float* refp  = (const float*)d_in[1];
  const float* feat  = (const float*)d_in[2];
  // d_in[3] spatial_shapes / d_in[4] level_start_index: constants (64x64, L=1)
  const float* ln_w  = (const float*)d_in[5];
  const float* ln_b  = (const float*)d_in[6];
  const float* vp_w  = (const float*)d_in[7];
  const float* vp_b  = (const float*)d_in[8];
  const float* off_w = (const float*)d_in[9];
  const float* off_b = (const float*)d_in[10];
  const float* aw_w  = (const float*)d_in[11];
  const float* aw_b  = (const float*)d_in[12];
  const float* out_w = (const float*)d_in[13];
  const float* out_b = (const float*)d_in[14];
  const float* gamma = (const float*)d_in[15];
  float* out = (float*)d_out;

  char* ws = (char*)d_ws;
  size_t off = 0;
  auto carve = [&](size_t bytes) -> char* {
    char* p = ws + off;
    off = (off + bytes + 255) & ~(size_t)255;
    return p;
  };
  u32*   q_pk    = (u32*)  carve((size_t)ROWS * KPACK * 4);
  u32*   feat_pk = (u32*)  carve((size_t)ROWS * KPACK * 4);
  float* value   = (float*)carve((size_t)ROWS * DIMX * 4);
  float* qo      = (float*)carve((size_t)ROWS * 72 * 4);
  u32*   attn_pk = (u32*)  carve((size_t)ROWS * KPACK * 4);
  u32*   wB_vp   = (u32*)  carve((size_t)DIMX * KPACK * 4);
  u32*   wB_oa   = (u32*)  carve((size_t)72 * KPACK * 4);
  u32*   wB_out  = (u32*)  carve((size_t)DIMX * KPACK * 4);
  float* bias72  = (float*)carve(72 * 4);

  // 1) q = LN(query) -> bf16 packed
  layernorm_pack<<<ROWS, 128, 0, stream>>>(query, ln_w, ln_b, q_pk);
  // 2) feat -> bf16 packed
  {
    int n = ROWS * KPACK;
    pack_rows<<<(n + 255) / 256, 256, 0, stream>>>(feat, feat_pk, n);
  }
  // 3) weights -> N-major bf16 packed B operands
  pack_weights<<<(DIMX * KPACK + 255) / 256, 256, 0, stream>>>(vp_w,  wB_vp, DIMX, 0);
  pack_weights<<<(48   * KPACK + 255) / 256, 256, 0, stream>>>(off_w, wB_oa, 48, 0);
  pack_weights<<<(24   * KPACK + 255) / 256, 256, 0, stream>>>(aw_w,  wB_oa, 24, 48);
  concat_bias<<<1, 128, 0, stream>>>(off_b, aw_b, bias72);

  // 4) value = feat @ vp_w + vp_b          (WMMA)
  gemm_bf16<<<dim3(DIMX / 128, ROWS / 128), 256, 0, stream>>>(
      feat_pk, wB_vp, vp_b, value, nullptr, nullptr, ROWS, DIMX, DIMX);
  // 5) qo = q @ [off_w | aw_w] + [off_b|aw_b]   (WMMA, N=72)
  gemm_bf16<<<dim3(1, ROWS / 128), 256, 0, stream>>>(
      q_pk, wB_oa, bias72, qo, nullptr, nullptr, ROWS, 72, DIMX);
  // 6) deformable sampling -> attn rows (bf16 packed)
  msda_sample<<<ROWS, 128, 0, stream>>>(value, qo, refp, attn_pk);
  // 7) out = query + gamma * (attn @ out_w + out_b)   (WMMA, fused epilogue)
  gemm_bf16<<<dim3(DIMX / 128, ROWS / 128), 256, 0, stream>>>(
      attn_pk, wB_out, out_b, out, query, gamma, ROWS, DIMX, DIMX);
}
